// Stage1Classifier_50087908606170
// MI455X (gfx1250) — matrix operand order repaired and checked by reading, not compiled, not run on hardware
//
#include <hip/hip_runtime.h>

typedef _Float16 half_t;
typedef __attribute__((ext_vector_type(16))) _Float16 v16h;
typedef __attribute__((ext_vector_type(8)))  _Float16 v8h;
typedef __attribute__((ext_vector_type(8)))  float    v8f;
typedef int v4i __attribute__((vector_size(16)));

#define NODES 8192
#define KNB   8
#define NEGS  0.01f

// ---------------------------------------------------------------------------
// Async global->LDS copy (CDNA5 GLOBAL_LOAD_ASYNC_TO_LDS_B128, ASYNCcnt).
// Builtin signature (from toolchain diagnostic): (v4i addrspace(1)*,
// v4i addrspace(3)*, imm offset, imm cpol). Falls back to a synchronous LDS
// store if the builtin is unavailable.
// ---------------------------------------------------------------------------
#if __has_builtin(__builtin_amdgcn_global_load_async_to_lds_b128)
#define ASYNC_LDS 1
#else
#define ASYNC_LDS 0
#endif

__device__ __forceinline__ void async_cp16(const half_t* gsrc, half_t* ldst) {
#if ASYNC_LDS
  __builtin_amdgcn_global_load_async_to_lds_b128(
      (__attribute__((address_space(1))) v4i*)gsrc,
      (__attribute__((address_space(3))) v4i*)ldst, 0, 0);
#else
  *(v8h*)ldst = *(const v8h*)gsrc;
#endif
}

__device__ __forceinline__ void async_wait0() {
#if ASYNC_LDS
#if __has_builtin(__builtin_amdgcn_s_wait_asynccnt)
  __builtin_amdgcn_s_wait_asynccnt(0);
#else
  asm volatile("s_wait_asynccnt 0x0" ::: "memory");
#endif
#endif
}

// ---------------------------------------------------------------------------
// Brute-force kNN (k=8) on first 3 coords. One thread per node, candidate
// positions staged through LDS in tiles of 256. Sorted 8-entry insertion list.
// ---------------------------------------------------------------------------
__global__ __launch_bounds__(256) void knn_kernel(const float* __restrict__ pos,
                                                  int ld, int* __restrict__ nbr) {
  __shared__ float sx[256], sy[256], sz[256];
  const int tid = threadIdx.x;
  const int i = blockIdx.x * 256 + tid;
  const float xi = pos[(size_t)i * ld + 0];
  const float yi = pos[(size_t)i * ld + 1];
  const float zi = pos[(size_t)i * ld + 2];
  float bd[KNB];
  int bi[KNB];
#pragma unroll
  for (int q = 0; q < KNB; ++q) { bd[q] = 3.0e38f; bi[q] = 0; }
  for (int base = 0; base < NODES; base += 256) {
    const int j = base + tid;
    sx[tid] = pos[(size_t)j * ld + 0];
    sy[tid] = pos[(size_t)j * ld + 1];
    sz[tid] = pos[(size_t)j * ld + 2];
    __syncthreads();
    for (int t = 0; t < 256; ++t) {
      const int jj = base + t;
      const float dx = sx[t] - xi, dy = sy[t] - yi, dz = sz[t] - zi;
      const float d = dx * dx + dy * dy + dz * dz;
      if (jj != i && d < bd[KNB - 1]) {
        int p = KNB - 1;
#pragma unroll
        for (int q = KNB - 2; q >= 0; --q) {
          if (d < bd[q]) { bd[q + 1] = bd[q]; bi[q + 1] = bi[q]; p = q; }
        }
        bd[p] = d; bi[p] = jj;
      }
    }
    __syncthreads();
  }
#pragma unroll
  for (int q = 0; q < KNB; ++q) nbr[i * KNB + q] = bi[q];
}

// ---------------------------------------------------------------------------
// WMMA GEMM: C[M,Ncp] = leaky(A[M,Kp] * B[Kp,Ncp] + bias), f16 in / f32 acc /
// f16 out. One wave -> 32x32 C tile (2x2 wmma). Block = 8 waves = 256 rows,
// all sharing one 32Kx32N B panel double-buffered in LDS via async DMA.
// grid = (Ncp/32, M/256). Fragment layouts per CDNA5 ISA 7.12.2.
// LDS B panel rows padded to 40 halves (80B) -> 16B aligned, low bank conflict.
// ---------------------------------------------------------------------------
#define BROW 40
__global__ __launch_bounds__(256) void gemm_bias_leaky(
    const half_t* __restrict__ A, int lda,
    const half_t* __restrict__ B, int ldb,
    const float* __restrict__ bias,
    half_t* __restrict__ C, int ldc, int Kp) {
  __shared__ half_t sB[2][32 * BROW];
  const int lane = threadIdx.x & 31;
  const int wave = threadIdx.x >> 5;
  const int n0 = blockIdx.x * 32;
  const int m0 = (blockIdx.y * 8 + wave) * 32;

  v8f acc00 = {}, acc01 = {}, acc10 = {}, acc11 = {};

  // A fragment addressing: lane 0-15 -> row M=lane, K 0-7 & 16-23;
  // lane 16-31 -> same rows, K 8-15 & 24-31.
  const int kb = (lane >> 4) * 8;
  const half_t* aptr0 = A + (size_t)(m0 + (lane & 15)) * lda + kb;
  const half_t* aptr1 = aptr0 + (size_t)16 * lda;

  const int steps = Kp >> 5;
  // Prologue: stage K-slab 0 into buffer 0 (wave 0 issues; lane = K row).
  if (wave == 0) {
    const half_t* bsrc = B + (size_t)lane * ldb + n0;
#pragma unroll
    for (int q = 0; q < 4; ++q)
      async_cp16(bsrc + q * 8, &sB[0][lane * BROW + q * 8]);
  }

  for (int s = 0; s < steps; ++s) {
    const int p = s & 1;
    if (wave == 0) async_wait0();
    __syncthreads();  // buf[p] ready; buf[p^1] fully consumed last step
    if (wave == 0 && s + 1 < steps) {
      const half_t* bsrc = B + (size_t)((s + 1) * 32 + lane) * ldb + n0;
#pragma unroll
      for (int q = 0; q < 4; ++q)
        async_cp16(bsrc + q * 8, &sB[p ^ 1][lane * BROW + q * 8]);
    }

    const int kk = s * 32;
    if (kk + 32 < Kp) __builtin_prefetch((const void*)(aptr0 + kk + 32), 0, 3);

    v8h a0lo = *(const v8h*)(aptr0 + kk);
    v8h a0hi = *(const v8h*)(aptr0 + kk + 16);
    v8h a1lo = *(const v8h*)(aptr1 + kk);
    v8h a1hi = *(const v8h*)(aptr1 + kk + 16);
    v16h a0, a1;
#pragma unroll
    for (int e = 0; e < 8; ++e) {
      a0[e] = a0lo[e]; a0[e + 8] = a0hi[e];
      a1[e] = a1lo[e]; a1[e + 8] = a1hi[e];
    }

    // B fragment from LDS: lane -> N column lane&15, K kb..kb+7 & 16+kb..16+kb+7.
    const half_t* bcol = &sB[p][kb * BROW + (lane & 15)];
    v16h b0, b1;
#pragma unroll
    for (int e = 0; e < 8; ++e) {
      b0[e] = bcol[e * BROW]; b0[e + 8] = bcol[(16 + e) * BROW];
      b1[e] = bcol[e * BROW + 16]; b1[e + 8] = bcol[(16 + e) * BROW + 16];
    }

    acc00 = __builtin_amdgcn_wmma_f32_16x16x32_f16(false, a0, false, b0,
                                                   (short)0, acc00, false, false);
    acc01 = __builtin_amdgcn_wmma_f32_16x16x32_f16(false, a0, false, b1,
                                                   (short)0, acc01, false, false);
    acc10 = __builtin_amdgcn_wmma_f32_16x16x32_f16(false, a1, false, b0,
                                                   (short)0, acc10, false, false);
    acc11 = __builtin_amdgcn_wmma_f32_16x16x32_f16(false, a1, false, b1,
                                                   (short)0, acc11, false, false);
  }

  // C/D layout: lane 0-15 -> N=lane, rows M=0..7 in VGPR r; lanes 16-31 -> M=8..15.
  const int nl = lane & 15;
  const int mr = (lane >> 4) * 8;
  const float bias0 = bias[n0 + nl];
  const float bias1 = bias[n0 + 16 + nl];
#pragma unroll
  for (int r = 0; r < 8; ++r) {
    float v00 = acc00[r] + bias0; v00 = (v00 >= 0.f) ? v00 : NEGS * v00;
    float v01 = acc01[r] + bias1; v01 = (v01 >= 0.f) ? v01 : NEGS * v01;
    float v10 = acc10[r] + bias0; v10 = (v10 >= 0.f) ? v10 : NEGS * v10;
    float v11 = acc11[r] + bias1; v11 = (v11 >= 0.f) ? v11 : NEGS * v11;
    C[(size_t)(m0 + mr + r) * ldc + n0 + nl] = (half_t)v00;
    C[(size_t)(m0 + mr + r) * ldc + n0 + 16 + nl] = (half_t)v01;
    C[(size_t)(m0 + 16 + mr + r) * ldc + n0 + nl] = (half_t)v10;
    C[(size_t)(m0 + 16 + mr + r) * ldc + n0 + 16 + nl] = (half_t)v11;
  }
}

// ---------------------------------------------------------------------------
// Build edge features E[i*K+k, :] = [x_i, x_j - x_i] (f16), zero K-padding.
// ---------------------------------------------------------------------------
__global__ __launch_bounds__(256) void build_edges_kernel(
    const float* __restrict__ h, int ld, int F, int Fp,
    const int* __restrict__ nbr, half_t* __restrict__ E) {
  const int idx = blockIdx.x * 256 + threadIdx.x;
  const int total = NODES * KNB * Fp;
  if (idx >= total) return;
  const int c = idx % Fp;
  const int r = idx / Fp;
  const int i = r >> 3;
  float v;
  if (c < F) {
    v = h[(size_t)i * ld + c];
  } else if (c < 2 * F) {
    const int j = nbr[r];
    const int cc = c - F;
    v = h[(size_t)j * ld + cc] - h[(size_t)i * ld + cc];
  } else {
    v = 0.f;
  }
  E[(size_t)idx] = (half_t)v;
}

// Sum over the 8 edges of each node; write f32 master + f16 into cat slot.
__global__ __launch_bounds__(256) void reduce_edges_kernel(
    const half_t* __restrict__ H2, float* __restrict__ hcur,
    half_t* __restrict__ catb, int catOff) {
  const int idx = blockIdx.x * 256 + threadIdx.x;
  if (idx >= NODES * 256) return;
  const int i = idx >> 8;
  const int c = idx & 255;
  float s = 0.f;
#pragma unroll
  for (int k = 0; k < KNB; ++k)
    s += (float)H2[(size_t)(i * KNB + k) * 256 + c];
  hcur[(size_t)i * 256 + c] = s;
  catb[(size_t)i * 1056 + catOff + c] = (half_t)s;
}

// cat[:,0:6] = x, cat[:,1030:1056] = 0
__global__ __launch_bounds__(256) void cat_x_kernel(const float* __restrict__ x,
                                                    half_t* __restrict__ catb) {
  const int idx = blockIdx.x * 256 + threadIdx.x;
  if (idx >= NODES * 32) return;
  const int i = idx >> 5;
  const int c = idx & 31;
  if (c < 6)
    catb[(size_t)i * 1056 + c] = (half_t)x[i * 6 + c];
  else
    catb[(size_t)i * 1056 + 1030 + (c - 6)] = (half_t)0.f;
}

// Pad-convert f32 weight [K0,N0] -> f16 [Kp,Np] (zeros outside).
__global__ __launch_bounds__(256) void pad_w_kernel(const float* __restrict__ src,
                                                    int K0, int N0,
                                                    half_t* __restrict__ dst,
                                                    int Kp, int Np) {
  const int idx = blockIdx.x * 256 + threadIdx.x;
  if (idx >= Kp * Np) return;
  const int k = idx / Np, n = idx % Np;
  const float v = (k < K0 && n < N0) ? src[k * N0 + n] : 0.f;
  dst[idx] = (half_t)v;
}

__global__ __launch_bounds__(256) void pad_b_kernel(const float* __restrict__ src,
                                                    int N0, float* __restrict__ dst,
                                                    int Np) {
  const int idx = blockIdx.x * 256 + threadIdx.x;
  if (idx >= Np) return;
  dst[idx] = (idx < N0) ? src[idx] : 0.f;
}

// logits[i] = g3[i,:] . Wh + bh   (no activation)
__global__ __launch_bounds__(256) void head_kernel(const half_t* __restrict__ g3,
                                                   const float* __restrict__ Wh,
                                                   const float* __restrict__ bh,
                                                   float* __restrict__ out) {
  const int i = blockIdx.x * 256 + threadIdx.x;
  if (i >= NODES) return;
  float s = bh[0];
#pragma unroll 8
  for (int c = 0; c < 128; ++c) s += (float)g3[(size_t)i * 128 + c] * Wh[c];
  out[i] = s;
}

// ---------------------------------------------------------------------------
extern "C" void kernel_launch(void* const* d_in, const int* in_sizes, int n_in,
                              void* d_out, int out_size, void* d_ws, size_t ws_size,
                              hipStream_t stream) {
  (void)in_sizes; (void)n_in; (void)out_size; (void)ws_size;
  const float* x = (const float*)d_in[0];
  const float* Wa[4] = {(const float*)d_in[1], (const float*)d_in[5],
                        (const float*)d_in[9], (const float*)d_in[13]};
  const float* ba[4] = {(const float*)d_in[2], (const float*)d_in[6],
                        (const float*)d_in[10], (const float*)d_in[14]};
  const float* Wb[4] = {(const float*)d_in[3], (const float*)d_in[7],
                        (const float*)d_in[11], (const float*)d_in[15]};
  const float* bb[4] = {(const float*)d_in[4], (const float*)d_in[8],
                        (const float*)d_in[12], (const float*)d_in[16]};
  const float* P1  = (const float*)d_in[17];
  const float* p1b = (const float*)d_in[18];
  const float* P2  = (const float*)d_in[19];
  const float* p2b = (const float*)d_in[20];
  const float* R   = (const float*)d_in[21];
  const float* rb  = (const float*)d_in[22];
  const float* Wh  = (const float*)d_in[23];
  const float* bh  = (const float*)d_in[24];
  float* out = (float*)d_out;

  // ---- workspace carve-out ----
  char* ws = (char*)d_ws;
  size_t off = 0;
  auto alloc = [&](size_t bytes) -> void* {
    void* p = ws + off;
    off += bytes;
    off = (off + 255) & ~(size_t)255;
    return p;
  };
  int*    nbr  = (int*)alloc((size_t)NODES * KNB * 4);
  float*  hcur = (float*)alloc((size_t)NODES * 256 * 4);
  half_t* catb = (half_t*)alloc((size_t)NODES * 1056 * 2);
  half_t* E    = (half_t*)alloc((size_t)NODES * KNB * 512 * 2);
  half_t* H1   = (half_t*)alloc((size_t)NODES * KNB * 352 * 2);
  half_t* H2   = E;  // alias: E dead once H1 is produced
  half_t* g1   = (half_t*)alloc((size_t)NODES * 352 * 2);
  half_t* g2   = (half_t*)alloc((size_t)NODES * 256 * 2);
  half_t* g3   = (half_t*)alloc((size_t)NODES * 128 * 2);
  half_t* WaP[4]; half_t* WbP[4]; float* baP[4]; float* bbP[4];
  WaP[0] = (half_t*)alloc((size_t)32 * 128 * 2);
  WbP[0] = (half_t*)alloc((size_t)128 * 256 * 2);
  for (int l = 1; l < 4; ++l) {
    WaP[l] = (half_t*)alloc((size_t)512 * 352 * 2);
    WbP[l] = (half_t*)alloc((size_t)352 * 256 * 2);
  }
  half_t* P1p = (half_t*)alloc((size_t)1056 * 352 * 2);
  half_t* P2p = (half_t*)alloc((size_t)352 * 256 * 2);
  half_t* Rp  = (half_t*)alloc((size_t)256 * 128 * 2);
  for (int l = 0; l < 4; ++l) {
    baP[l] = (float*)alloc(352 * 4);
    bbP[l] = (float*)alloc(256 * 4);
  }
  float* p1bp = (float*)alloc(352 * 4);
  float* p2bp = (float*)alloc(256 * 4);
  float* rbp  = (float*)alloc(128 * 4);

  auto cdiv = [](long a, long b) { return (int)((a + b - 1) / b); };

  // ---- weight/bias conversion (f32 -> f16, zero-padded) ----
  pad_w_kernel<<<cdiv(32 * 128, 256), 256, 0, stream>>>(Wa[0], 12, 128, WaP[0], 32, 128);
  pad_b_kernel<<<1, 256, 0, stream>>>(ba[0], 128, baP[0], 128);
  pad_w_kernel<<<cdiv(128 * 256, 256), 256, 0, stream>>>(Wb[0], 128, 256, WbP[0], 128, 256);
  pad_b_kernel<<<1, 256, 0, stream>>>(bb[0], 256, bbP[0], 256);
  for (int l = 1; l < 4; ++l) {
    pad_w_kernel<<<cdiv(512 * 352, 256), 256, 0, stream>>>(Wa[l], 512, 336, WaP[l], 512, 352);
    pad_b_kernel<<<2, 256, 0, stream>>>(ba[l], 336, baP[l], 352);
    pad_w_kernel<<<cdiv(352 * 256, 256), 256, 0, stream>>>(Wb[l], 336, 256, WbP[l], 352, 256);
    pad_b_kernel<<<1, 256, 0, stream>>>(bb[l], 256, bbP[l], 256);
  }
  pad_w_kernel<<<cdiv(1056 * 352, 256), 256, 0, stream>>>(P1, 1030, 336, P1p, 1056, 352);
  pad_b_kernel<<<2, 256, 0, stream>>>(p1b, 336, p1bp, 352);
  pad_w_kernel<<<cdiv(352 * 256, 256), 256, 0, stream>>>(P2, 336, 256, P2p, 352, 256);
  pad_b_kernel<<<1, 256, 0, stream>>>(p2b, 256, p2bp, 256);
  pad_w_kernel<<<cdiv(256 * 128, 256), 256, 0, stream>>>(R, 256, 128, Rp, 256, 128);
  pad_b_kernel<<<1, 256, 0, stream>>>(rb, 128, rbp, 128);

  cat_x_kernel<<<cdiv((long)NODES * 32, 256), 256, 0, stream>>>(x, catb);

  const int ME = NODES * KNB;  // 65536 edge rows

  // ---- initial graph on xyz of x ----
  knn_kernel<<<NODES / 256, 256, 0, stream>>>(x, 6, nbr);

  // ---- conv layer 1: 6 -> (2F=12 pad 32) -> 128 -> 256 ----
  build_edges_kernel<<<cdiv((long)ME * 32, 256), 256, 0, stream>>>(x, 6, 6, 32, nbr, E);
  gemm_bias_leaky<<<dim3(128 / 32, ME / 256), 256, 0, stream>>>(E, 32, WaP[0], 128, baP[0], H1, 352, 32);
  gemm_bias_leaky<<<dim3(256 / 32, ME / 256), 256, 0, stream>>>(H1, 352, WbP[0], 256, bbP[0], H2, 256, 128);
  reduce_edges_kernel<<<cdiv((long)NODES * 256, 256), 256, 0, stream>>>(H2, hcur, catb, 6);
  knn_kernel<<<NODES / 256, 256, 0, stream>>>(hcur, 256, nbr);

  // ---- conv layers 2..4: 256 -> 512 -> 336(pad 352) -> 256 ----
  for (int l = 1; l < 4; ++l) {
    build_edges_kernel<<<cdiv((long)ME * 512, 256), 256, 0, stream>>>(hcur, 256, 256, 512, nbr, E);
    gemm_bias_leaky<<<dim3(352 / 32, ME / 256), 256, 0, stream>>>(E, 512, WaP[l], 352, baP[l], H1, 352, 512);
    gemm_bias_leaky<<<dim3(256 / 32, ME / 256), 256, 0, stream>>>(H1, 352, WbP[l], 256, bbP[l], H2, 256, 352);
    reduce_edges_kernel<<<cdiv((long)NODES * 256, 256), 256, 0, stream>>>(H2, hcur, catb, 6 + 256 * l);
    if (l < 3) knn_kernel<<<NODES / 256, 256, 0, stream>>>(hcur, 256, nbr);
  }

  // ---- readout MLPs: 1030(pad 1056) -> 336(pad 352) -> 256 -> 128 -> 1 ----
  gemm_bias_leaky<<<dim3(352 / 32, NODES / 256), 256, 0, stream>>>(catb, 1056, P1p, 352, p1bp, g1, 352, 1056);
  gemm_bias_leaky<<<dim3(256 / 32, NODES / 256), 256, 0, stream>>>(g1, 352, P2p, 256, p2bp, g2, 256, 352);
  gemm_bias_leaky<<<dim3(128 / 32, NODES / 256), 256, 0, stream>>>(g2, 256, Rp, 128, rbp, g3, 128, 256);
  head_kernel<<<NODES / 256, 256, 0, stream>>>(g3, Wh, bh, out);
}
